// DeepseekV2MLAAttention_17403207483761
// MI455X (gfx1250) — compile-verified, matrix operand order checked by reading, not run on hardware
//
#include <hip/hip_runtime.h>
#include <hip/hip_bf16.h>
#include <math.h>

typedef __bf16 bf16;
typedef __attribute__((ext_vector_type(16))) __bf16 v16bf;
typedef __attribute__((ext_vector_type(8)))  __bf16 v8bf;
typedef __attribute__((ext_vector_type(8)))  float  v8f;

#define S_LEN 2048
#define E_DIM 2048
#define H_N   16
#define DN_   128
#define DR_   64
#define R_DIM 512
#define QLR_  1536
#define DQ    (R_DIM + DR_)   // 576 : attention head dim (latent 512 + rope 64)
#define HD    (DN_ + DR_)     // 192 : raw q head dim
#define QB    (H_N * HD)      // 3072
#define OFD   (H_N * R_DIM)   // 8192 : flattened latent-output dim
#define EPS_F 1e-6f

__device__ __forceinline__ int imin_(int a, int b) { return a < b ? a : b; }

__device__ __forceinline__ v8f zero_v8f() {
  v8f z;
#pragma unroll
  for (int i = 0; i < 8; ++i) z[i] = 0.f;
  return z;
}

// Load a 16x32 bf16 WMMA operand fragment (A layout; BT rows for the NT gemm).
// colbase points at the k-origin column; `row` is the per-lane row index.
// CDNA5 16-bit A layout: lane<16 holds K {0..7,16..23} of row lane,
// lane>=16 holds K {8..15,24..31} of row lane-16. Two 16B contiguous loads.
__device__ __forceinline__ v16bf load_frag16x32(const bf16* colbase, int ld, int row) {
  int lane = threadIdx.x & 31;
  const bf16* p = colbase + (size_t)row * ld + ((lane >> 4) << 3);
  v8bf lo = *reinterpret_cast<const v8bf*>(p);
  v8bf hi = *reinterpret_cast<const v8bf*>(p + 16);
  v16bf f;
#pragma unroll
  for (int i = 0; i < 8; ++i) { f[i] = lo[i]; f[i + 8] = hi[i]; }
  return f;
}

__device__ __forceinline__ v16bf wmma_bf16(v16bf a, v16bf b, v8f& acc);

// Async copy of one 128x32 bf16 B tile into LDS (8 KB) by a 256-thread block.
// thread t: tile row = t>>1, 16-element chunk = (t&1)*16; one
// global_load_async_to_lds_b128 moves 16B (INST_OFFSET applies to both sides).
__device__ __forceinline__ void async_fill_bt(const bf16* BT, int ldb, int n_base,
                                              int N, int k0, bf16* dst_lds) {
  int tid = threadIdx.x;
  int row = tid >> 1;
  int chunk = (tid & 1) << 4;
  int brow = imin_(n_base + row, N - 1);
  const bf16* g = BT + (size_t)brow * ldb + k0 + chunk;
  unsigned lds = (unsigned)(uintptr_t)(dst_lds + row * 32 + chunk);   // LDS aperture: low 32 bits
  unsigned long long ga = (unsigned long long)(uintptr_t)g;
  asm volatile("global_load_async_to_lds_b128 %0, %1, off"
               :: "v"(lds), "v"(ga) : "memory");
}

__device__ __forceinline__ void wait_async_lds() {
  asm volatile("s_wait_asynccnt 0x0" ::: "memory");
}

// ---------------------------------------------------------------- prep kernels

__global__ __launch_bounds__(256) void cast_bf16_kernel(const float* __restrict__ src,
                                                        bf16* __restrict__ dst, int n) {
  int i = blockIdx.x * 256 + threadIdx.x;
  if (i < n) dst[i] = (bf16)src[i];
}

// dst[cols][rows] = (bf16) src[rows][cols]
__global__ __launch_bounds__(256) void transpose_cast_kernel(const float* __restrict__ src,
                                                             bf16* __restrict__ dst,
                                                             int rows, int cols) {
  __shared__ float tile[32][33];
  int r0 = blockIdx.y * 32, c0 = blockIdx.x * 32;
  int tx = threadIdx.x, ty = threadIdx.y;  // (32, 8)
#pragma unroll
  for (int i = 0; i < 32; i += 8) {
    int r = r0 + ty + i, c = c0 + tx;
    if (r < rows && c < cols) tile[ty + i][tx] = src[(size_t)r * cols + c];
  }
  __syncthreads();
#pragma unroll
  for (int i = 0; i < 32; i += 8) {
    int r = r0 + tx, c = c0 + ty + i;
    if (r < rows && c < cols) dst[(size_t)c * rows + r] = (bf16)tile[tx][ty + i];
  }
}

// split w_kv_b [512, H*256] into w_uk [H][512][128] and w_uv [H][512][128] (bf16)
__global__ __launch_bounds__(256) void extract_kvb_kernel(const float* __restrict__ wkvb,
                                                          bf16* __restrict__ uk,
                                                          bf16* __restrict__ uv) {
  int idx = blockIdx.x * 256 + threadIdx.x;
  if (idx >= R_DIM * H_N * 256) return;
  int r = idx >> 12;
  int c = idx & 4095;
  int h = c >> 8;
  int d = c & 255;
  float v = wkvb[idx];
  if (d < 128) uk[((size_t)h * R_DIM + r) * 128 + d] = (bf16)v;
  else         uv[((size_t)h * R_DIM + r) * 128 + (d - 128)] = (bf16)v;
}

__global__ __launch_bounds__(256) void rmsnorm_cast_kernel(const float* __restrict__ src,
                                                           const float* __restrict__ gamma,
                                                           bf16* __restrict__ dst,
                                                           int C, int lds, int ldd) {
  int row = blockIdx.x, tid = threadIdx.x;
  const float* s = src + (size_t)row * lds;
  __shared__ float red[256];
  float a = 0.f;
  for (int c = tid; c < C; c += 256) { float v = s[c]; a += v * v; }
  red[tid] = a;
  __syncthreads();
  for (int off = 128; off > 0; off >>= 1) {
    if (tid < off) red[tid] += red[tid + off];
    __syncthreads();
  }
  float rms = rsqrtf(red[0] / (float)C + EPS_F);
  for (int c = tid; c < C; c += 256)
    dst[(size_t)row * ldd + c] = (bf16)(s[c] * rms * gamma[c]);
}

// per row s: rmsnorm first 512 -> k_full[:, :512] and vT (transposed), rope last 64
__global__ __launch_bounds__(256) void kv_prep_kernel(const float* __restrict__ ckvkpe,
                                                      const float* __restrict__ gamma,
                                                      bf16* __restrict__ kfull,
                                                      bf16* __restrict__ vT) {
  int s = blockIdx.x, tid = threadIdx.x;
  const float* row = ckvkpe + (size_t)s * DQ;
  __shared__ float red[256];
  float a = 0.f;
  for (int r = tid; r < R_DIM; r += 256) { float v = row[r]; a += v * v; }
  red[tid] = a;
  __syncthreads();
  for (int off = 128; off > 0; off >>= 1) {
    if (tid < off) red[tid] += red[tid + off];
    __syncthreads();
  }
  float rms = rsqrtf(red[0] / (float)R_DIM + EPS_F);
  for (int r = tid; r < R_DIM; r += 256) {
    float v = row[r] * rms * gamma[r];
    kfull[(size_t)s * DQ + r] = (bf16)v;
    vT[(size_t)r * S_LEN + s] = (bf16)v;
  }
  if (tid < DR_) {
    int fi = tid & 31;
    float inv = __powf(10000.f, -(float)fi / 32.f);
    float ang = (float)s * inv;
    float c = __cosf(ang), sn = __sinf(ang);
    const float* pe = row + R_DIM;
    float x2 = (tid < 32) ? -pe[tid + 32] : pe[tid - 32];
    kfull[(size_t)s * DQ + R_DIM + tid] = (bf16)(pe[tid] * c + x2 * sn);
  }
}

// per (s,h): cast q_nope -> q_nope_bf [S][H*128]; rope q_pe -> q_full[h][s][512..576]
__global__ __launch_bounds__(128) void q_prep_kernel(const float* __restrict__ qraw,
                                                     bf16* __restrict__ qnope,
                                                     bf16* __restrict__ qfull) {
  int s = blockIdx.x, h = blockIdx.y, tid = threadIdx.x;
  const float* base = qraw + (size_t)s * QB + h * HD;
  qnope[(size_t)s * (H_N * DN_) + h * DN_ + tid] = (bf16)base[tid];
  if (tid < DR_) {
    int fi = tid & 31;
    float inv = __powf(10000.f, -(float)fi / 32.f);
    float ang = (float)s * inv;
    float c = __cosf(ang), sn = __sinf(ang);
    const float* pe = base + DN_;
    float x2 = (tid < 32) ? -pe[tid + 32] : pe[tid - 32];
    qfull[((size_t)h * S_LEN + s) * DQ + R_DIM + tid] = (bf16)(pe[tid] * c + x2 * sn);
  }
}

// -------------------------------------------------------------- WMMA NT GEMM
// C[M,N] = A[M,K] * BT[N,K]^T, bf16 in, f32 accumulate, f32 or bf16 out.
// 256 threads = 8 waves: block tile 64 (M) x 128 (N); wave tile 16 x 64.
// B tile (128x32, shared by the 4 wm-waves) is staged into LDS with
// global_load_async_to_lds_b128, double-buffered (ASYNCcnt + barrier).
template <bool BF16OUT>
__global__ __launch_bounds__(256) void gemm_nt_kernel(const bf16* __restrict__ A,
                                                      const bf16* __restrict__ BT,
                                                      void* __restrict__ Cv,
                                                      int M, int N, int K,
                                                      int lda, int ldb, int ldc) {
  __shared__ __align__(16) bf16 s_bt[2][128 * 32];   // 2 x 8 KB stages
  int tid = threadIdx.x;
  int lane = tid & 31;
  int w = tid >> 5;
  int wm = w >> 1, wn = w & 1;
  int m0 = blockIdx.y * 64 + wm * 16;
  int n_blk = blockIdx.x * 128;
  int n0 = n_blk + wn * 64;
  int lrow = lane & 15;
  int arow = imin_(m0 + lrow, M - 1);

  v8f acc[4];
#pragma unroll
  for (int t = 0; t < 4; ++t) acc[t] = zero_v8f();

  // prologue: stage 0
  async_fill_bt(BT, ldb, n_blk, N, 0, s_bt[0]);
  wait_async_lds();
  __syncthreads();

  v16bf a_cur = load_frag16x32(A, lda, arow);
  int buf = 0;

  for (int k0 = 0; k0 < K; k0 += 32) {
    int knext = k0 + 32;
    bool more = knext < K;
    if (more) {
      async_fill_bt(BT, ldb, n_blk, N, knext, s_bt[buf ^ 1]);
      __builtin_prefetch(A + (size_t)arow * lda + knext, 0, 0);
    }
    v16bf a_next = a_cur;
    if (more) a_next = load_frag16x32(A + knext, lda, arow);

    // double-buffered B fragments from LDS (ds_load_b128 path)
    v16bf b_cur = load_frag16x32(s_bt[buf], 32, wn * 64 + lrow);
#pragma unroll
    for (int t = 0; t < 4; ++t) {
      v16bf b_next = b_cur;
      if (t < 3) b_next = load_frag16x32(s_bt[buf], 32, wn * 64 + (t + 1) * 16 + lrow);
      acc[t] = __builtin_amdgcn_wmma_f32_16x16x32_bf16(false, a_cur, false, b_cur,
                                                       (short)0, acc[t], false, false);
      b_cur = b_next;
    }
    a_cur = a_next;
    if (more) wait_async_lds();
    __syncthreads();
    buf ^= 1;
  }

  int rowb = (lane >> 4) << 3;
#pragma unroll
  for (int t = 0; t < 4; ++t) {
#pragma unroll
    for (int j = 0; j < 8; ++j) {
      int r = m0 + rowb + j;
      int c = n0 + t * 16 + lrow;
      if (r < M && c < N) {
        if (BF16OUT) ((bf16*)Cv)[(size_t)r * ldc + c] = (bf16)acc[t][j];
        else         ((float*)Cv)[(size_t)r * ldc + c] = acc[t][j];
      }
    }
  }
}

// ------------------------------------------------------ flash MLA attention
// grid (S/16, H), block 128 = 4 waves. Each block: 16 q rows of head h.
// wave w: score subtile cols [w*16, w*16+16) of a 64-wide k tile,
//         output cols [w*128, w*128+128) of the 512-dim latent V.
__global__ __launch_bounds__(128) void mla_attn_kernel(const bf16* __restrict__ qfull,
                                                       const bf16* __restrict__ kfull,
                                                       const bf16* __restrict__ vT,
                                                       bf16* __restrict__ ofl) {
  const int S = S_LEN;
  int q0 = blockIdx.x * 16;
  int h  = blockIdx.y;
  int tid = threadIdx.x;
  int w = tid >> 5;
  int lane = tid & 31;
  int lrow = lane & 15;
  int rowb = (lane >> 4) << 3;

  __shared__ __align__(16) float s_sc[16 * 64];
  __shared__ __align__(16) bf16  s_p[16 * 64];
  __shared__ float s_m[16], s_l[16], s_a[16], s_red[16 * 8];

  const bf16* qbase = qfull + ((size_t)h * S + q0) * DQ;
  v16bf qf[DQ / 32];
#pragma unroll
  for (int kq = 0; kq < DQ / 32; ++kq)
    qf[kq] = load_frag16x32(qbase + kq * 32, DQ, lrow);

  v8f acc[8];
#pragma unroll
  for (int t = 0; t < 8; ++t) acc[t] = zero_v8f();

  if (tid < 16) { s_m[tid] = -1e30f; s_l[tid] = 0.f; }
  __syncthreads();

  const float scale = 0.07216878364870322f;  // (DN+DR)^-0.5 = 192^-0.5

  for (int k0 = 0; k0 < q0 + 16; k0 += 64) {   // causal upper bound; S%64==0
    int kb = k0 + w * 16;
    // ---- scores: S = Q Kt  (18 WMMA over K=576), K frags double-buffered
    v8f sf = zero_v8f();
    v16bf k_cur = load_frag16x32(kfull, DQ, kb + lrow);
#pragma unroll
    for (int kq = 0; kq < DQ / 32; ++kq) {
      v16bf k_next = k_cur;
      if (kq < DQ / 32 - 1)
        k_next = load_frag16x32(kfull + (kq + 1) * 32, DQ, kb + lrow);
      sf = __builtin_amdgcn_wmma_f32_16x16x32_bf16(false, qf[kq], false, k_cur,
                                                   (short)0, sf, false, false);
      k_cur = k_next;
    }
#pragma unroll
    for (int j = 0; j < 8; ++j) {
      int rq = q0 + rowb + j;
      int ck = kb + lrow;
      float v = sf[j] * scale;
      if (ck > rq) v = -1e30f;                // causal mask
      s_sc[(rowb + j) * 64 + w * 16 + lrow] = v;
    }
    __syncthreads();

    // ---- online softmax over the 16x64 tile (8 threads per row)
    {
      int r = tid >> 3, t8 = tid & 7;
      float lm = -1e30f;
#pragma unroll
      for (int c = 0; c < 8; ++c) lm = fmaxf(lm, s_sc[r * 64 + t8 * 8 + c]);
      s_red[r * 8 + t8] = lm;
    }
    __syncthreads();
    if ((tid & 7) == 0) {
      int r = tid >> 3;
      float m = s_red[r * 8];
      for (int i = 1; i < 8; ++i) m = fmaxf(m, s_red[r * 8 + i]);
      float mold = s_m[r];
      float mnew = fmaxf(mold, m);
      s_a[r] = __expf(mold - mnew);
      s_m[r] = mnew;
    }
    __syncthreads();
    {
      int r = tid >> 3, t8 = tid & 7;
      float mnew = s_m[r];
      float ls = 0.f;
#pragma unroll
      for (int c = 0; c < 8; ++c) {
        float sc = s_sc[r * 64 + t8 * 8 + c];
        float e = (sc <= -1e29f) ? 0.f : __expf(sc - mnew);
        s_p[r * 64 + t8 * 8 + c] = (bf16)e;
        ls += e;
      }
      s_red[r * 8 + t8] = ls;
    }
    __syncthreads();
    if ((tid & 7) == 0) {
      int r = tid >> 3;
      float sum = 0.f;
      for (int i = 0; i < 8; ++i) sum += s_red[r * 8 + i];
      s_l[r] = s_l[r] * s_a[r] + sum;
    }
    __syncthreads();

    // ---- rescale running output, then O += P Vt (16 WMMA, V double-buffered)
#pragma unroll
    for (int t = 0; t < 8; ++t)
#pragma unroll
      for (int j = 0; j < 8; ++j) acc[t][j] *= s_a[rowb + j];

#pragma unroll
    for (int kk = 0; kk < 64; kk += 32) {
      v16bf pf = load_frag16x32(s_p + kk, 64, lrow);
      v16bf v_cur = load_frag16x32(vT + (k0 + kk), S, w * 128 + lrow);
#pragma unroll
      for (int t = 0; t < 8; ++t) {
        v16bf v_next = v_cur;
        if (t < 7)
          v_next = load_frag16x32(vT + (k0 + kk), S, w * 128 + (t + 1) * 16 + lrow);
        acc[t] = __builtin_amdgcn_wmma_f32_16x16x32_bf16(false, pf, false, v_cur,
                                                         (short)0, acc[t], false, false);
        v_cur = v_next;
      }
    }
    __syncthreads();
  }

  // ---- normalize, write o_flat [S][H*512] bf16
#pragma unroll
  for (int t = 0; t < 8; ++t) {
#pragma unroll
    for (int j = 0; j < 8; ++j) {
      int r = rowb + j;
      float v = acc[t][j] / s_l[r];
      int col = w * 128 + t * 16 + lrow;
      ofl[(size_t)(q0 + r) * OFD + h * R_DIM + col] = (bf16)v;
    }
  }
}

// ------------------------------------------------------------------ launcher

extern "C" void kernel_launch(void* const* d_in, const int* in_sizes, int n_in,
                              void* d_out, int out_size, void* d_ws, size_t ws_size,
                              hipStream_t stream) {
  (void)in_sizes; (void)n_in; (void)out_size; (void)ws_size;
  const float* x      = (const float*)d_in[0];
  const float* w_q_a  = (const float*)d_in[1];
  const float* q_ln   = (const float*)d_in[2];
  const float* w_q_b  = (const float*)d_in[3];
  const float* w_kv_a = (const float*)d_in[4];
  const float* kv_ln  = (const float*)d_in[5];
  const float* w_kv_b = (const float*)d_in[6];
  const float* w_o    = (const float*)d_in[7];

  char* p = (char*)d_ws;
  auto carve = [&](size_t elems, size_t esz) {
    void* r = (void*)p;
    p += (elems * esz + 255) & ~(size_t)255;
    return r;
  };

  bf16*  x_bf    = (bf16*)carve((size_t)S_LEN * E_DIM, 2);
  bf16*  wqaT    = (bf16*)carve((size_t)QLR_ * E_DIM, 2);          // [1536][2048]
  bf16*  wqbT    = (bf16*)carve((size_t)QB * QLR_, 2);             // [3072][1536]
  bf16*  wkvaT   = (bf16*)carve((size_t)DQ * E_DIM, 2);            // [576][2048]
  bf16*  woT     = (bf16*)carve((size_t)E_DIM * (H_N * 128), 2);   // [2048][2048]
  bf16*  wuk     = (bf16*)carve((size_t)H_N * R_DIM * DN_, 2);     // [16][512][128]
  bf16*  wuv     = (bf16*)carve((size_t)H_N * R_DIM * DN_, 2);     // [16][512][128]
  float* qaraw   = (float*)carve((size_t)S_LEN * QLR_, 4);
  bf16*  qabf    = (bf16*)carve((size_t)S_LEN * QLR_, 2);
  float* qraw    = (float*)carve((size_t)S_LEN * QB, 4);
  float* ckvkpe  = (float*)carve((size_t)S_LEN * DQ, 4);
  bf16*  kfull   = (bf16*)carve((size_t)S_LEN * DQ, 2);            // [S][576]
  bf16*  vT      = (bf16*)carve((size_t)R_DIM * S_LEN, 2);         // [512][S]
  bf16*  qnope   = (bf16*)carve((size_t)S_LEN * (H_N * DN_), 2);   // [S][2048]
  bf16*  qfull   = (bf16*)carve((size_t)H_N * S_LEN * DQ, 2);      // [H][S][576]
  bf16*  woabsT  = (bf16*)carve((size_t)E_DIM * OFD, 2);           // [2048][8192]
  bf16*  ofl     = (bf16*)carve((size_t)S_LEN * OFD, 2);           // [S][8192]

  // 1) cast x to bf16
  {
    int n = S_LEN * E_DIM;
    cast_bf16_kernel<<<(n + 255) / 256, 256, 0, stream>>>(x, x_bf, n);
  }
  // 2) transpose+cast weights into NT (N x K) bf16 layouts
  {
    dim3 b(32, 8);
    transpose_cast_kernel<<<dim3((QLR_ + 31) / 32, (E_DIM + 31) / 32), b, 0, stream>>>(w_q_a, wqaT, E_DIM, QLR_);
    transpose_cast_kernel<<<dim3((QB + 31) / 32, (QLR_ + 31) / 32), b, 0, stream>>>(w_q_b, wqbT, QLR_, QB);
    transpose_cast_kernel<<<dim3((DQ + 31) / 32, (E_DIM + 31) / 32), b, 0, stream>>>(w_kv_a, wkvaT, E_DIM, DQ);
    transpose_cast_kernel<<<dim3((E_DIM + 31) / 32, (H_N * 128 + 31) / 32), b, 0, stream>>>(w_o, woT, H_N * 128, E_DIM);
  }
  {
    int n = R_DIM * H_N * 256;
    extract_kvb_kernel<<<(n + 255) / 256, 256, 0, stream>>>(w_kv_b, wuk, wuv);
  }
  // 3) q_a = x @ w_q_a   (f32 out), then rmsnorm -> bf16
  gemm_nt_kernel<false><<<dim3(QLR_ / 128, S_LEN / 64), 256, 0, stream>>>(
      x_bf, wqaT, qaraw, S_LEN, QLR_, E_DIM, E_DIM, E_DIM, QLR_);
  rmsnorm_cast_kernel<<<S_LEN, 256, 0, stream>>>(qaraw, q_ln, qabf, QLR_, QLR_, QLR_);
  // 4) q = q_a @ w_q_b   (f32 out)
  gemm_nt_kernel<false><<<dim3(QB / 128, S_LEN / 64), 256, 0, stream>>>(
      qabf, wqbT, qraw, S_LEN, QB, QLR_, QLR_, QLR_, QB);
  // 5) ckv_kpe = x @ w_kv_a (f32 out, N=576 -> bounds-checked tail tile)
  gemm_nt_kernel<false><<<dim3((DQ + 127) / 128, S_LEN / 64), 256, 0, stream>>>(
      x_bf, wkvaT, ckvkpe, S_LEN, DQ, E_DIM, E_DIM, E_DIM, DQ);
  // 6) k_full / vT (rmsnorm + rope), q_nope/q_pe (cast + rope)
  kv_prep_kernel<<<S_LEN, 256, 0, stream>>>(ckvkpe, kv_ln, kfull, vT);
  q_prep_kernel<<<dim3(S_LEN, H_N), 128, 0, stream>>>(qraw, qnope, qfull);
  // 7) per head: q_abs = q_nope_h @ w_uk_h^T  -> q_full[h][:, :512] (bf16 out)
  //    and w_o_abs^T_h = w_o_h^T @ w_uv_h^T   -> woabsT[:, h*512..] (bf16 out)
  for (int h = 0; h < H_N; ++h) {
    gemm_nt_kernel<true><<<dim3(R_DIM / 128, S_LEN / 64), 256, 0, stream>>>(
        qnope + h * DN_, wuk + (size_t)h * R_DIM * DN_,
        qfull + (size_t)h * S_LEN * DQ,
        S_LEN, R_DIM, DN_, H_N * DN_, DN_, DQ);
    gemm_nt_kernel<true><<<dim3(R_DIM / 128, E_DIM / 64), 256, 0, stream>>>(
        woT + h * 128, wuv + (size_t)h * R_DIM * DN_,
        woabsT + h * R_DIM,
        E_DIM, R_DIM, DN_, E_DIM, DN_, OFD);
  }
  // 8) causal flash attention in latent space
  mla_attn_kernel<<<dim3(S_LEN / 16, H_N), 128, 0, stream>>>(qfull, kfull, vT, ofl);
  // 9) y = o_flat @ w_o_abs  (one big GEMM, K = H*512 = 8192, f32 out)
  gemm_nt_kernel<false><<<dim3(E_DIM / 128, S_LEN / 64), 256, 0, stream>>>(
      ofl, woabsT, (float*)d_out, S_LEN, E_DIM, OFD, OFD, OFD, E_DIM);
}
